// CharacterLanguageModel_50869592654494
// MI455X (gfx1250) — matrix operand order repaired and checked by reading, not compiled, not run on hardware
//
#include <hip/hip_runtime.h>
#include <hip/hip_bf16.h>

// Problem dims
#define BB   64
#define TT   1024
#define VV   100
#define EE   100
#define HH   1024
#define G4H  4096      // 4*H
#define EP   128       // E padded to mult of 32 for WMMA K
#define VP   112       // V padded to mult of 16 for WMMA N

// Persistent LSTM kernel shape: 64 blocks x 128 threads = 256 waves.
// Wave w owns gate-col tile ct = w (0..255) and computes ALL 4 batch row
// tiles (4 accumulators sharing each B tile -> Whh streamed exactly once
// per step). h (128KB bf16) is staged into LDS per block per step by the
// Tensor Data Mover, overlapped with the x-projection WMMAs.
#define NBLK 64
#define NTHR 128
#define LDSW 1032      // padded LDS row stride (bf16): 2064B = 129*16B, bank-rotating

typedef __attribute__((ext_vector_type(16))) __bf16   v16bf;
typedef __attribute__((ext_vector_type(8)))  float    v8f;
typedef __attribute__((ext_vector_type(4)))  unsigned u32x4;
typedef __attribute__((ext_vector_type(4)))  float    f32x4;
typedef __attribute__((ext_vector_type(4)))  int      i32x4;
typedef __attribute__((ext_vector_type(8)))  int      i32x8;

union BF16Tile { u32x4 u[2]; v16bf v; };

static __device__ __forceinline__ __bf16 f2bf(float f) {
  union { float f; unsigned u; } x; x.f = f;
  unsigned r = x.u + 0x7FFFu + ((x.u >> 16) & 1u);   // round-to-nearest-even
  unsigned short h = (unsigned short)(r >> 16);
  __bf16 o; __builtin_memcpy(&o, &h, 2); return o;
}

static __device__ __forceinline__ float sigmoidf_(float x) {
  return 1.0f / (1.0f + __expf(-x));
}

// ---- software grid barrier (all NBLK blocks resident by construction) ----
static __device__ __forceinline__ void grid_barrier(unsigned* cnt, unsigned* gen) {
  __syncthreads();
  if (threadIdx.x == 0) {
    __threadfence();
    unsigned g = __atomic_load_n(gen, __ATOMIC_ACQUIRE);
    unsigned a = __atomic_fetch_add(cnt, 1u, __ATOMIC_ACQ_REL);
    if (a == NBLK - 1u) {
      __atomic_store_n(cnt, 0u, __ATOMIC_RELAXED);
      __atomic_store_n(gen, g + 1u, __ATOMIC_RELEASE);
    } else {
      while (__atomic_load_n(gen, __ATOMIC_ACQUIRE) == g) {
        __builtin_amdgcn_s_sleep(1);
      }
    }
  }
  __syncthreads();
}

// ---- TDM: DMA h [64 x 1024] bf16 global -> LDS with bank padding ----
// D# per CDNA5 ISA 8.3/8.4: 2-D tensor, data_size=2B,
// tensor_dim0=1024, tensor_dim1=64, stride0=1024,
// tile_dim0=1024, tile_dim1=64,
// pad_enable, pad_interval=7 (256 DWORDs = 1024B), pad_amount=1 (2 DWORDs = 8B)
//   -> LDS row stride 2048+16 = 2064B (= LDSW bf16), +8B shift at element 512.
#if defined(__has_builtin)
#if __has_builtin(__builtin_amdgcn_tensor_load_to_lds)
#define HAVE_TDM 1
#endif
#endif

static __device__ __forceinline__ void tdm_stage_h(const __bf16* src, unsigned ldsOff) {
#ifdef HAVE_TDM
  unsigned long long ga = (unsigned long long)(size_t)src;
  u32x4 g0;
  g0[0] = 1u;                                          // count=1, user descriptor
  g0[1] = ldsOff;                                      // lds_addr (bytes)
  g0[2] = (unsigned)(ga & 0xFFFFFFFFu);                // global_addr[31:0]
  g0[3] = (unsigned)((ga >> 32) & 0x01FFFFFFu)         // global_addr[56:32]
        | (2u << 30);                                  // type = 2 ("image")
  i32x8 g1;
  g1[0] = (int)((1u << 16)                             // data_size = 2 bytes
              | (1u << 20)                             // pad_enable
              | (7u << 22)                             // pad_interval: 256 DWORDs
              | (1u << 25));                           // pad_amount: 2 DWORDs
  g1[1] = (int)(((unsigned)HH & 0xFFFFu) << 16);       // tensor_dim0[15:0] = 1024
  g1[2] = (int)(((unsigned)BB & 0xFFFFu) << 16);       // dim0[31:16]=0, tensor_dim1[15:0]=64
  g1[3] = (int)(((unsigned)HH & 0xFFFFu) << 16);       // dim1[31:16]=0, tile_dim0 = 1024
  g1[4] = (int)((unsigned)BB & 0xFFFFu);               // tile_dim1 = 64, tile_dim2 = 0
  g1[5] = (int)HH;                                     // tensor_dim0_stride[31:0] = 1024
  g1[6] = 0;                                           // stride0[47:32]=0, stride1 lo=0
  g1[7] = 0;
  i32x4 gz = {0, 0, 0, 0};
#if __has_include(<hip/amd_detail/amd_gfx1250_TDM.h>)
  i32x8 gz8 = {0, 0, 0, 0, 0, 0, 0, 0};                // clang-23 / therock: 6-arg form
  __builtin_amdgcn_tensor_load_to_lds(g0, g1, gz, gz, gz8, 0);
#else
  __builtin_amdgcn_tensor_load_to_lds(g0, g1, gz, gz, 0);   // ROCm 7.2: 5-arg form
#endif
#else
  (void)src; (void)ldsOff;
#endif
}

// ================= prep kernels =================

__global__ void k_pad_bf16(const float* __restrict__ src, __bf16* __restrict__ dst,
                           int srcR, int srcC, int dstR, int dstC) {
  int idx = blockIdx.x * blockDim.x + threadIdx.x;
  int n = dstR * dstC;
  if (idx >= n) return;
  int r = idx / dstC, c = idx % dstC;
  float v = (r < srcR && c < srcC) ? src[(size_t)r * srcC + c] : 0.0f;
  dst[idx] = f2bf(v);
}

__global__ void k_bias(const float* __restrict__ bih, const float* __restrict__ bhh,
                       float* __restrict__ bias) {
  int i = blockIdx.x * blockDim.x + threadIdx.x;
  if (i < G4H) bias[i] = bih[i] + bhh[i];
}

// embedding gather -> bf16 [B*T, EP] (zero-padded cols 100..127)
__global__ void k_embed(const int* __restrict__ chars, const float* __restrict__ emb,
                        __bf16* __restrict__ x) {
  int idx = blockIdx.x * blockDim.x + threadIdx.x;
  if (idx >= BB * TT * EP) return;
  int bt = idx / EP, e = idx - bt * EP;
  float v = (e < EE) ? emb[(size_t)chars[bt] * EE + e] : 0.0f;
  x[idx] = f2bf(v);
}

__global__ void k_init_state(float* __restrict__ hf, float* __restrict__ cf,
                             __bf16* __restrict__ hb0, __bf16* __restrict__ hb1) {
  int i = blockIdx.x * blockDim.x + threadIdx.x;
  if (i >= BB * HH) return;
  hf[i] = 0.0f; cf[i] = 0.0f;
  hb0[i] = f2bf(0.0f); hb1[i] = f2bf(0.0f);
}

// ================= persistent LSTM =================

__global__ void __launch_bounds__(NTHR)
k_lstm(const int* __restrict__ charlens,
       const __bf16* __restrict__ x,      // [B*T, EP]
       const __bf16* __restrict__ Wih,    // [4H, EP]  bf16
       const __bf16* __restrict__ Whh,    // [4H, H]   bf16
       const float*  __restrict__ bias,   // [4H] = bih+bhh
       float* __restrict__ hf, float* __restrict__ cf,   // [B*H] f32 state
       __bf16* __restrict__ hb0, __bf16* __restrict__ hb1, // h bf16 double-buffer
       float* __restrict__ gbuf,          // [B, 4H] per-step gate preacts
       unsigned* __restrict__ bar,        // {cnt, gen}
       float* __restrict__ out)           // d_out: output | hn | cn | decoded
{
  extern __shared__ char smem_raw[];
  __bf16* hsh = (__bf16*)smem_raw;       // [BB][LDSW] staged h, TDM bank-padded

  const int lane = threadIdx.x & 31;
  const int wid  = threadIdx.x >> 5;     // wave in block
  const int ct   = blockIdx.x * (NTHR / 32) + wid;   // 0..255
  const int arow = lane & 15;            // A-matrix row within tile (bf16 16x32 layout)
  const int ahk  = (lane >> 4) * 8;      // A-matrix K base
  const int bcol = lane & 15;            // B/C col within tile
  const int bhk  = (lane >> 4) * 16;     // unused: B loads use ahk (same k order as A)
  const int m0   = (lane >> 4) * 8;      // C-matrix row base
  const float biasv = bias[ct * 16 + bcol];
  const unsigned ldsOff = (unsigned)(size_t)(void*)hsh;  // LDS aperture: low 32 bits

  unsigned* cnt = bar;
  unsigned* gen = bar + 1;

  const int tid = blockIdx.x * NTHR + threadIdx.x; // 0..8191 (phase B)

  for (int t = 0; t < TT; ++t) {
    const __bf16* hcur  = (t & 1) ? hb1 : hb0;
    __bf16*       hnext = (t & 1) ? hb0 : hb1;

#ifdef HAVE_TDM
    // ---------- async stage h into LDS via Tensor Data Mover ----------
    if (wid == 0) tdm_stage_h(hcur, ldsOff);
#else
    {
      const int c8 = threadIdx.x * 8;
      #pragma unroll 8
      for (int j = 0; j < BB; ++j) {
        int pad4 = 0;                                  // mirror TDM pad layout
        u32x4 v = *(const u32x4*)(hcur + (size_t)j * HH + c8);
        if (c8 >= 512) pad4 = 4;
        *(u32x4*)(hsh + (size_t)j * LDSW + c8 + pad4) = v;
      }
    }
#endif

    v8f acc[4];
    #pragma unroll
    for (int rt = 0; rt < 4; ++rt)
      #pragma unroll
      for (int i = 0; i < 8; ++i) acc[rt][i] = biasv;

    // ---------- x projection (K=0..127, global A) overlaps the TDM DMA ----------
    {
      const __bf16* wr = Wih + (size_t)(ct * 16 + bcol) * EP;
      #pragma unroll
      for (int k0 = 0; k0 < EP; k0 += 32) {
        BF16Tile b;
        b.u[0] = *(const u32x4*)(wr + k0 + ahk);
        b.u[1] = *(const u32x4*)(wr + k0 + ahk + 16);
        #pragma unroll
        for (int rt = 0; rt < 4; ++rt) {
          const __bf16* xr = x + ((size_t)(rt * 16 + arow) * TT + t) * EP + k0 + ahk;
          BF16Tile a;
          a.u[0] = *(const u32x4*)(xr);
          a.u[1] = *(const u32x4*)(xr + 16);
          acc[rt] = __builtin_amdgcn_wmma_f32_16x16x32_bf16(false, a.v, false, b.v,
                                                            (short)0, acc[rt], false, false);
        }
      }
    }

    // prefetch next step's embedding rows while the big GEMM runs
    if (t + 1 < TT) {
      #pragma unroll
      for (int rt = 0; rt < 4; ++rt)
        __builtin_prefetch(x + ((size_t)(rt * 16 + arow) * TT + (t + 1)) * EP, 0, 3);
    }

#ifdef HAVE_TDM
    if (wid == 0) __builtin_amdgcn_s_wait_tensorcnt(0);  // DMA complete
#endif
    __syncthreads();                                     // LDS h visible to all waves

    // ---------- recurrent projection: K=0..1023; B streamed once, A from LDS ----------
    {
      const __bf16* wh = Whh + (size_t)(ct * 16 + bcol) * HH;
      #pragma unroll 4
      for (int k0 = 0; k0 < HH; k0 += 32) {
        const int pad4 = (k0 >= 512) ? 4 : 0;            // TDM pad shift for 2nd half-row
        BF16Tile b;
        b.u[0] = *(const u32x4*)(wh + k0 + ahk);
        b.u[1] = *(const u32x4*)(wh + k0 + ahk + 16);
        #pragma unroll
        for (int rt = 0; rt < 4; ++rt) {
          const __bf16* hr = hsh + (size_t)(rt * 16 + arow) * LDSW + k0 + ahk + pad4;
          BF16Tile a;
          a.u[0] = *(const u32x4*)(hr);
          a.u[1] = *(const u32x4*)(hr + 16);
          acc[rt] = __builtin_amdgcn_wmma_f32_16x16x32_bf16(false, a.v, false, b.v,
                                                            (short)0, acc[rt], false, false);
        }
      }
    }

    #pragma unroll
    for (int rt = 0; rt < 4; ++rt) {
      float* gr = gbuf + (size_t)(rt * 16 + m0) * G4H + ct * 16 + bcol;
      #pragma unroll
      for (int i = 0; i < 8; ++i) gr[(size_t)i * G4H] = acc[rt][i];
    }

    grid_barrier(cnt, gen);

    // ---------- phase B: gates + state update (packed-seq masking) ----------
    {
      #pragma unroll
      for (int rep = 0; rep < 8; ++rep) {
        int idx = tid + rep * (NBLK * NTHR);     // 0..65535 == B*H
        int b  = idx >> 10;
        int hh = idx & (HH - 1);
        const float* gb = gbuf + (size_t)b * G4H + hh;
        float i_ = sigmoidf_(gb[0]);
        float f_ = sigmoidf_(gb[HH]);
        float g_ = tanhf(gb[2 * HH]);
        float o_ = sigmoidf_(gb[3 * HH]);
        float cprev = cf[idx];
        float cnew  = f_ * cprev + i_ * g_;
        float hnew  = o_ * tanhf(cnew);
        bool valid = t < charlens[b];
        float hkeep = valid ? hnew : hf[idx];
        float ckeep = valid ? cnew : cprev;
        hf[idx] = hkeep;
        cf[idx] = ckeep;
        hnext[idx] = f2bf(hkeep);
        out[((size_t)b * TT + t) * HH + hh] = valid ? hnew : 0.0f;
      }
    }

    grid_barrier(cnt, gen);
  }

  // ---------- hn, cn ----------
  #pragma unroll
  for (int rep = 0; rep < 8; ++rep) {
    int idx = tid + rep * (NBLK * NTHR);
    out[(size_t)BB * TT * HH + idx]            = hf[idx];
    out[(size_t)BB * TT * HH + BB * HH + idx]  = cf[idx];
  }
}

// ================= decode GEMM: decoded = output * Wdec^T + bdec =================

#define DEC_WPB 4
__global__ void __launch_bounds__(128)
k_decode(const float* __restrict__ outp,   // [B*T, H] f32 (LSTM outputs, zeros past len)
         const __bf16* __restrict__ Wdec,  // [VP, H] bf16, rows 100..111 zero
         const float*  __restrict__ bdec,  // [V]
         float* __restrict__ dec)          // [B*T, V]
{
  const int lane = threadIdx.x & 31;
  const int wave = blockIdx.x * DEC_WPB + (threadIdx.x >> 5);
  const int NT = VP / 16;                 // 7 col tiles
  const int mt = wave / NT;               // 0..4095
  const int ct = wave - mt * NT;          // 0..6
  const int arow = lane & 15;
  const int ahk  = (lane >> 4) * 8;
  const int bcol = lane & 15;
  const int bhk  = (lane >> 4) * 16;
  const int m0   = (lane >> 4) * 8;
  const int col  = ct * 16 + bcol;

  v8f acc;
  float biasv = (col < VV) ? bdec[col] : 0.0f;
  #pragma unroll
  for (int i = 0; i < 8; ++i) acc[i] = biasv;

  const float*  ar = outp + (size_t)(mt * 16 + arow) * HH;
  const __bf16* br = Wdec + (size_t)col * HH;

  #pragma unroll 4
  for (int k0 = 0; k0 < HH; k0 += 32) {
    BF16Tile a, b;
    #pragma unroll
    for (int g = 0; g < 2; ++g) {        // f32 -> bf16 in-register for A
      const float* p = ar + k0 + ahk + g * 16;
      f32x4 lo = *(const f32x4*)(p);
      f32x4 hi = *(const f32x4*)(p + 4);
      __bf16 tmp[8] = { f2bf(lo[0]), f2bf(lo[1]), f2bf(lo[2]), f2bf(lo[3]),
                        f2bf(hi[0]), f2bf(hi[1]), f2bf(hi[2]), f2bf(hi[3]) };
      __builtin_memcpy(&a.u[g], tmp, 16);
    }
    b.u[0] = *(const u32x4*)(br + k0 + ahk);
    b.u[1] = *(const u32x4*)(br + k0 + ahk + 16);
    acc = __builtin_amdgcn_wmma_f32_16x16x32_bf16(false, a.v, false, b.v,
                                                  (short)0, acc, false, false);
  }

  if (col < VV) {
    float* dr = dec + (size_t)(mt * 16 + m0) * VV + col;
    #pragma unroll
    for (int i = 0; i < 8; ++i) dr[(size_t)i * VV] = acc[i];
  }
}

// ================= launcher =================

// workspace layout (bytes), all regions 16B-aligned
constexpr size_t OFF_CNT  = 0;                                   // 256 B barrier
constexpr size_t OFF_BIAS = 256;                                 // 4096 f32
constexpr size_t OFF_WIH  = OFF_BIAS + (size_t)G4H * 4;          // [4H,EP] bf16
constexpr size_t OFF_WHH  = OFF_WIH  + (size_t)G4H * EP * 2;     // [4H,H]  bf16
constexpr size_t OFF_WDEC = OFF_WHH  + (size_t)G4H * HH * 2;     // [VP,H]  bf16
constexpr size_t OFF_X    = OFF_WDEC + (size_t)VP * HH * 2;      // [B*T,EP] bf16
constexpr size_t OFF_HF   = OFF_X    + (size_t)BB * TT * EP * 2; // [B*H] f32
constexpr size_t OFF_CF   = OFF_HF   + (size_t)BB * HH * 4;
constexpr size_t OFF_HB0  = OFF_CF   + (size_t)BB * HH * 4;      // [B*H] bf16
constexpr size_t OFF_HB1  = OFF_HB0  + (size_t)BB * HH * 2;
constexpr size_t OFF_G    = OFF_HB1  + (size_t)BB * HH * 2;      // [B,4H] f32

extern "C" void kernel_launch(void* const* d_in, const int* in_sizes, int n_in,
                              void* d_out, int out_size, void* d_ws, size_t ws_size,
                              hipStream_t stream) {
  const int*   chars    = (const int*)d_in[0];
  const int*   charlens = (const int*)d_in[1];
  const float* emb      = (const float*)d_in[2];
  const float* Wih      = (const float*)d_in[3];
  const float* Whh      = (const float*)d_in[4];
  const float* bih      = (const float*)d_in[5];
  const float* bhh      = (const float*)d_in[6];
  const float* Wdec     = (const float*)d_in[7];
  const float* bdec     = (const float*)d_in[8];
  float* out = (float*)d_out;

  char* ws = (char*)d_ws;
  unsigned* bar   = (unsigned*)(ws + OFF_CNT);
  float*    bias  = (float*)   (ws + OFF_BIAS);
  __bf16*   wihb  = (__bf16*)  (ws + OFF_WIH);
  __bf16*   whhb  = (__bf16*)  (ws + OFF_WHH);
  __bf16*   wdecb = (__bf16*)  (ws + OFF_WDEC);
  __bf16*   xb    = (__bf16*)  (ws + OFF_X);
  float*    hf    = (float*)   (ws + OFF_HF);
  float*    cf    = (float*)   (ws + OFF_CF);
  __bf16*   hb0   = (__bf16*)  (ws + OFF_HB0);
  __bf16*   hb1   = (__bf16*)  (ws + OFF_HB1);
  float*    gbuf  = (float*)   (ws + OFF_G);

  hipMemsetAsync(ws + OFF_CNT, 0, 256, stream);

  k_bias<<<(G4H + 255) / 256, 256, 0, stream>>>(bih, bhh, bias);
  {
    int n = G4H * EP;
    k_pad_bf16<<<(n + 255) / 256, 256, 0, stream>>>(Wih, wihb, G4H, EE, G4H, EP);
  }
  {
    int n = G4H * HH;
    k_pad_bf16<<<(n + 255) / 256, 256, 0, stream>>>(Whh, whhb, G4H, HH, G4H, HH);
  }
  {
    int n = VP * HH;
    k_pad_bf16<<<(n + 255) / 256, 256, 0, stream>>>(Wdec, wdecb, VV, HH, VP, HH);
  }
  {
    int n = BB * TT * EP;
    k_embed<<<(n + 255) / 256, 256, 0, stream>>>(chars, emb, xb);
  }
  {
    int n = BB * HH;
    k_init_state<<<(n + 255) / 256, 256, 0, stream>>>(hf, cf, hb0, hb1);
  }

  // 64 blocks, 128 threads, 64*1032*2 = 132096 B dynamic LDS (h staged, bank-padded)
  k_lstm<<<NBLK, NTHR, (size_t)BB * LDSW * 2, stream>>>(
      charlens, xb, wihb, whhb, bias, hf, cf, hb0, hb1, gbuf, bar, out);

  {
    const size_t dec_off = (size_t)BB * TT * HH + 2ull * BB * HH;
    int ntiles = (BB * TT / 16) * (VP / 16);       // 4096 * 7
    k_decode<<<ntiles / DEC_WPB, 128, 0, stream>>>(out, wdecb, bdec, out + dec_off);
  }
}
